// GriffinAnomalyDetector_55284819034484
// MI455X (gfx1250) — matrix operand order repaired and checked by reading, not compiled, not run on hardware
//
#include <hip/hip_runtime.h>
#include <math.h>

#define D_MODEL 512
#define D_INNER 1024
#define D_MLP   2048
#define NLAYERS 4
#define BB      16
#define TT      1024
#define MROWS   (BB*TT)          /* 16384 token rows */
#define LOG8F   2.0794415416798357f

typedef __attribute__((ext_vector_type(16))) __bf16 v16bf;
typedef __attribute__((ext_vector_type(8)))  __bf16 v8bf;
typedef __attribute__((ext_vector_type(8)))  float  v8f;

#define ACT_NONE 0
#define ACT_GELU 1

__device__ __forceinline__ __bf16 to_bf(float f) {
  union { float f; unsigned u; } v; v.f = f;
  unsigned r = (v.u + 0x7FFFu + ((v.u >> 16) & 1u)) >> 16;   // RNE
  unsigned short s = (unsigned short)r;
  __bf16 o; __builtin_memcpy(&o, &s, 2); return o;
}

__device__ __forceinline__ float gelu_exact(float x) {
  return 0.5f * x * (1.0f + erff(x * 0.70710678118654752f));
}

// ---------------------------------------------------------------------------
// bf16 WMMA GEMM:  C[M,N] = act(A[M,K] @ B[K,N] + bias) + resid
// B is supplied transposed (Bt[N,K], row-major) so B fragments are contiguous.
// Block = 256 threads (8 waves); block tile 128(M) x 64(N); wave tile 16x64.
// ---------------------------------------------------------------------------
__global__ __launch_bounds__(256) void gemm_bf16_wmma(
    const __bf16* __restrict__ A, const __bf16* __restrict__ Bt,
    const float* __restrict__ bias, const float* __restrict__ resid,
    float* __restrict__ outF, __bf16* __restrict__ outB,
    int M, int N, int K, int act)
{
  const int lane = threadIdx.x & 31;
  const int wave = threadIdx.x >> 5;
  const int half = lane >> 4;       // 0: lanes 0-15, 1: lanes 16-31
  const int idx  = lane & 15;
  const int tm = blockIdx.y * 128 + wave * 16;
  const int tn = blockIdx.x * 64;

  const __bf16* arow  = A  + (size_t)(tm + idx) * K;
  const __bf16* brow0 = Bt + (size_t)(tn +  0 + idx) * K;
  const __bf16* brow1 = Bt + (size_t)(tn + 16 + idx) * K;
  const __bf16* brow2 = Bt + (size_t)(tn + 32 + idx) * K;
  const __bf16* brow3 = Bt + (size_t)(tn + 48 + idx) * K;

  v8f acc0 = {}, acc1 = {}, acc2 = {}, acc3 = {};

  for (int k = 0; k < K; k += 32) {
    // A fragment (16x32 bf16): lane half 0 -> K k..k+15, half 1 -> k+16..k+31
    const int ka = k + half * 16;
    v8bf a_lo = *(const v8bf*)(arow + ka);
    v8bf a_hi = *(const v8bf*)(arow + ka + 8);
    v16bf av = __builtin_shufflevector(a_lo, a_hi,
        0,1,2,3,4,5,6,7,8,9,10,11,12,13,14,15);
    // B fragment (32x16 bf16): lane half 0 -> K k..k+15, half 1 -> k+16..k+31
    const int kb = k + half * 16;
    v16bf b0 = *(const v16bf*)(brow0 + kb);
    v16bf b1 = *(const v16bf*)(brow1 + kb);
    v16bf b2 = *(const v16bf*)(brow2 + kb);
    v16bf b3 = *(const v16bf*)(brow3 + kb);
    acc0 = __builtin_amdgcn_wmma_f32_16x16x32_bf16(false, av, false, b0, (short)0, acc0, false, false);
    acc1 = __builtin_amdgcn_wmma_f32_16x16x32_bf16(false, av, false, b1, (short)0, acc1, false, false);
    acc2 = __builtin_amdgcn_wmma_f32_16x16x32_bf16(false, av, false, b2, (short)0, acc2, false, false);
    acc3 = __builtin_amdgcn_wmma_f32_16x16x32_bf16(false, av, false, b3, (short)0, acc3, false, false);
  }

  v8f accs[4] = {acc0, acc1, acc2, acc3};
  #pragma unroll
  for (int j = 0; j < 4; ++j) {
    const int n = tn + j * 16 + idx;
    const float bv = bias ? bias[n] : 0.0f;
    #pragma unroll
    for (int r = 0; r < 8; ++r) {
      const int m = tm + r + 8 * half;        // C/D layout: VGPR r, lane half
      float v = accs[j][r] + bv;
      if (act == ACT_GELU) v = gelu_exact(v);
      size_t o = (size_t)m * N + n;
      if (resid) v += resid[o];
      if (outF) outF[o] = v;
      if (outB) outB[o] = to_bf(v);
    }
  }
}

// ---------------------------------------------------------------------------
// LayerNorm over last dim (512); one wave32 per row; emits bf16 for WMMA.
// ---------------------------------------------------------------------------
__global__ __launch_bounds__(256) void layernorm_to_bf16(
    const float* __restrict__ X, const float* __restrict__ g,
    const float* __restrict__ b, __bf16* __restrict__ out, int M)
{
  const int lane = threadIdx.x & 31;
  const int wave = threadIdx.x >> 5;
  const int row  = blockIdx.x * 8 + wave;
  if (row >= M) return;
  const float* xr = X + (size_t)row * D_MODEL;
  float vals[16];
  float s = 0.0f;
  #pragma unroll
  for (int j = 0; j < 16; ++j) { vals[j] = xr[lane + j * 32]; s += vals[j]; }
  #pragma unroll
  for (int m = 16; m >= 1; m >>= 1) s += __shfl_xor(s, m, 32);
  const float mu = s * (1.0f / 512.0f);
  float var = 0.0f;
  #pragma unroll
  for (int j = 0; j < 16; ++j) { float d = vals[j] - mu; var += d * d; }
  #pragma unroll
  for (int m = 16; m >= 1; m >>= 1) var += __shfl_xor(var, m, 32);
  const float rstd = rsqrtf(var * (1.0f / 512.0f) + 1e-5f);
  __bf16* orow = out + (size_t)row * D_MODEL;
  #pragma unroll
  for (int j = 0; j < 16; ++j) {
    const int c = lane + j * 32;
    orow[c] = to_bf((vals[j] - mu) * rstd * g[c] + b[c]);
  }
}

// ---------------------------------------------------------------------------
// Causal depthwise conv, kernel 4, left pad 3; emits bf16 for the RG-LRU GEMMs.
// ---------------------------------------------------------------------------
__global__ __launch_bounds__(256) void dwconv_to_bf16(
    const float* __restrict__ R, const float* __restrict__ w,
    const float* __restrict__ bias, __bf16* __restrict__ out)
{
  const unsigned gid = blockIdx.x * blockDim.x + threadIdx.x;  // < 16*1024*1024
  const int d = gid & (D_INNER - 1);
  const int t = (gid >> 10) & (TT - 1);
  const int b = gid >> 20;
  if (b >= BB) return;
  float acc = bias[d];
  #pragma unroll
  for (int j = 0; j < 4; ++j) {
    const int tt = t - 3 + j;
    if (tt >= 0) acc += R[((size_t)b * TT + tt) * D_INNER + d] * w[d * 4 + j];
  }
  out[gid] = to_bf(acc);
}

// ---------------------------------------------------------------------------
// RG-LRU: fused sigmoid/softplus gates + diagonal scan over T + gelu-gate mul.
// One thread per (batch, channel); 16384 independent recurrences.
// ---------------------------------------------------------------------------
__global__ __launch_bounds__(256) void rglru_scan(
    const float* __restrict__ rp, const float* __restrict__ ip,
    const float* __restrict__ xw, const float* __restrict__ g,
    __bf16* __restrict__ rg)
{
  const int gid = blockIdx.x * blockDim.x + threadIdx.x;  // < 16384
  const int d = gid & (D_INNER - 1);
  const int b = gid >> 10;
  if (b >= BB) return;
  float h = 0.0f;
  for (int t = 0; t < TT; ++t) {
    const size_t o = ((size_t)b * TT + t) * D_INNER + d;
    const float r = 1.0f / (1.0f + expf(-rp[o]));
    const float z = LOG8F * r;
    const float sp = (z > 0.0f) ? (z + log1pf(expf(-z))) : log1pf(expf(z));
    const float a = expf(-sp);
    const float gate = sqrtf(fmaxf(1.0f - a * a, 1e-6f));
    const float i = 1.0f / (1.0f + expf(-ip[o]));
    const float u = gate * i * xw[o];
    h = a * h + u;
    rg[o] = to_bf(h * g[o]);
  }
}

// ---------------------------------------------------------------------------
// Tiny K projections (K=7 in, N=7 out): plain VALU kernels, not worth WMMA.
// ---------------------------------------------------------------------------
__global__ __launch_bounds__(256) void input_proj(
    const float* __restrict__ x, const float* __restrict__ w,
    const float* __restrict__ b, float* __restrict__ X)
{
  const unsigned gid = blockIdx.x * blockDim.x + threadIdx.x;  // M*512
  const int n = gid & (D_MODEL - 1);
  const int m = gid >> 9;
  if (m >= MROWS) return;
  float acc = b[n];
  #pragma unroll
  for (int i = 0; i < 7; ++i) acc += x[(size_t)m * 7 + i] * w[i * D_MODEL + n];
  X[gid] = acc;
}

__global__ __launch_bounds__(256) void output_proj(
    const float* __restrict__ X, const float* __restrict__ w,
    const float* __restrict__ b, float* __restrict__ out)
{
  const unsigned gid = blockIdx.x * blockDim.x + threadIdx.x;  // M*7
  if (gid >= MROWS * 7u) return;
  const int i = gid % 7;
  const int m = gid / 7;
  float acc = b[i];
  for (int d = 0; d < D_MODEL; ++d)
    acc += X[(size_t)m * D_MODEL + d] * w[d * 7 + i];
  out[gid] = acc;
}

// fp32 (K,N) weight -> bf16 transposed (N,K) for contiguous B fragments
__global__ __launch_bounds__(256) void convert_transpose(
    const float* __restrict__ W, __bf16* __restrict__ Wt, int K, int N)
{
  const unsigned gid = blockIdx.x * blockDim.x + threadIdx.x;
  if (gid >= (unsigned)(K * N)) return;
  const int k = gid / N;
  const int n = gid % N;
  Wt[(size_t)n * K + k] = to_bf(W[gid]);
}

// ---------------------------------------------------------------------------
extern "C" void kernel_launch(void* const* d_in, const int* in_sizes, int n_in,
                              void* d_out, int out_size, void* d_ws, size_t ws_size,
                              hipStream_t stream) {
  (void)in_sizes; (void)n_in; (void)out_size; (void)ws_size;

  const float* x_in  = (const float*)d_in[0];
  const float* in_w  = (const float*)d_in[1];
  const float* in_b  = (const float*)d_in[2];
  const float* out_w = (const float*)d_in[3];
  const float* out_b = (const float*)d_in[4];
  auto P = [&](int L, int j) -> const float* { return (const float*)d_in[5 + 18 * L + j]; };
  // per-layer param slots: 0 ln1_g 1 ln1_b 2 ln2_g 3 ln2_b 4 in_proj 5 conv_w
  // 6 conv_b 7 Wr 8 br 9 Wi 10 bi 11 Wx 12 gate_proj 13 out_proj
  // 14 mlp_w1 15 mlp_b1 16 mlp_w2 17 mlp_b2

  char* ws = (char*)d_ws;
  size_t off = 0;
  auto alloc = [&](size_t bytes) -> void* {
    void* p = ws + off; off += (bytes + 255) & ~(size_t)255; return p;
  };

  float*  X  = (float*) alloc((size_t)MROWS * D_MODEL * 4);   // residual stream
  __bf16* XN = (__bf16*)alloc((size_t)MROWS * D_MODEL * 2);   // LN output (bf16)
  float*  F1 = (float*) alloc((size_t)MROWS * D_INNER * 4);   // r_pre / xw / mlp-hid(bf16 alias)
  float*  F2 = (float*) alloc((size_t)MROWS * D_INNER * 4);   // gelu gate g
  float*  F3 = (float*) alloc((size_t)MROWS * D_INNER * 4);   // rp
  float*  F4 = (float*) alloc((size_t)MROWS * D_INNER * 4);   // ip
  __bf16* H1 = (__bf16*)alloc((size_t)MROWS * D_INNER * 2);   // conv out / rg (bf16)
  __bf16* HB = (__bf16*)F1;                                   // mlp hidden alias (bf16)

  // bf16 transposed weights, per layer: inT gateT WrT WiT WxT outT w1T w2T
  __bf16* WT[NLAYERS][8];
  const int WK[8] = {512, 512, 1024, 1024, 1024, 1024, 512, 2048};
  const int WN[8] = {1024, 1024, 1024, 1024, 1024, 512, 2048, 512};
  const int WSLOT[8] = {4, 12, 7, 9, 11, 13, 14, 16};
  for (int L = 0; L < NLAYERS; ++L)
    for (int j = 0; j < 8; ++j)
      WT[L][j] = (__bf16*)alloc((size_t)WK[j] * WN[j] * 2);

  for (int L = 0; L < NLAYERS; ++L)
    for (int j = 0; j < 8; ++j) {
      const int total = WK[j] * WN[j];
      convert_transpose<<<(total + 255) / 256, 256, 0, stream>>>(
          P(L, WSLOT[j]), WT[L][j], WK[j], WN[j]);
    }

  // h = x @ in_w + in_b
  input_proj<<<(MROWS * D_MODEL) / 256, 256, 0, stream>>>(x_in, in_w, in_b, X);

  const dim3 blk(256);
  const dim3 gridN1024(D_INNER / 64, MROWS / 128);
  const dim3 gridN512(D_MODEL / 64, MROWS / 128);
  const dim3 gridN2048(D_MLP / 64, MROWS / 128);

  for (int L = 0; L < NLAYERS; ++L) {
    // LN1 -> XN (bf16)
    layernorm_to_bf16<<<MROWS / 8, blk, 0, stream>>>(X, P(L, 0), P(L, 1), XN, MROWS);
    // r_pre = XN @ in_proj               (fp32 -> F1)
    gemm_bf16_wmma<<<gridN1024, blk, 0, stream>>>(XN, WT[L][0], nullptr, nullptr,
        F1, nullptr, MROWS, D_INNER, D_MODEL, ACT_NONE);
    // g = gelu(XN @ gate_proj)           (fp32 -> F2)
    gemm_bf16_wmma<<<gridN1024, blk, 0, stream>>>(XN, WT[L][1], nullptr, nullptr,
        F2, nullptr, MROWS, D_INNER, D_MODEL, ACT_GELU);
    // causal depthwise conv -> H1 (bf16)
    dwconv_to_bf16<<<(MROWS * D_INNER) / 256, blk, 0, stream>>>(F1, P(L, 5), P(L, 6), H1);
    // rp = conv @ Wr + br ; ip = conv @ Wi + bi ; xw = conv @ Wx
    gemm_bf16_wmma<<<gridN1024, blk, 0, stream>>>(H1, WT[L][2], P(L, 8), nullptr,
        F3, nullptr, MROWS, D_INNER, D_INNER, ACT_NONE);
    gemm_bf16_wmma<<<gridN1024, blk, 0, stream>>>(H1, WT[L][3], P(L, 10), nullptr,
        F4, nullptr, MROWS, D_INNER, D_INNER, ACT_NONE);
    gemm_bf16_wmma<<<gridN1024, blk, 0, stream>>>(H1, WT[L][4], nullptr, nullptr,
        F1, nullptr, MROWS, D_INNER, D_INNER, ACT_NONE);
    // RG-LRU scan + multiply by gelu gate -> rg (bf16, reuses H1)
    rglru_scan<<<(BB * D_INNER) / 256, blk, 0, stream>>>(F3, F4, F1, F2, H1);
    // X = X + rg @ out_proj
    gemm_bf16_wmma<<<gridN512, blk, 0, stream>>>(H1, WT[L][5], nullptr, X,
        X, nullptr, MROWS, D_MODEL, D_INNER, ACT_NONE);
    // LN2 -> XN
    layernorm_to_bf16<<<MROWS / 8, blk, 0, stream>>>(X, P(L, 2), P(L, 3), XN, MROWS);
    // hid = gelu(XN @ mlp_w1 + b1)       (bf16 -> HB)
    gemm_bf16_wmma<<<gridN2048, blk, 0, stream>>>(XN, WT[L][6], P(L, 15), nullptr,
        nullptr, HB, MROWS, D_MLP, D_MODEL, ACT_GELU);
    // X = X + hid @ mlp_w2 + b2
    gemm_bf16_wmma<<<gridN512, blk, 0, stream>>>(HB, WT[L][7], P(L, 17), X,
        X, nullptr, MROWS, D_MODEL, D_MLP, ACT_NONE);
  }

  // pred = X @ out_w + out_b
  output_proj<<<(MROWS * 7 + 255) / 256, 256, 0, stream>>>(X, out_w, out_b, (float*)d_out);
}
